// LCNNConvolution_5111011082748
// MI455X (gfx1250) — matrix operand order repaired and checked by reading, not compiled, not run on hardware
//
#include <hip/hip_runtime.h>
#include <hip/hip_fp16.h>

typedef _Float16 half16 __attribute__((ext_vector_type(16)));
typedef _Float16 half8  __attribute__((ext_vector_type(8)));
typedef float    v8f    __attribute__((ext_vector_type(8)));

#define N_SITES 50000
#define NODE_F  64
#define N_PERM  12
#define N_NEIGH 8
#define IN_F    512
#define OUT_F   64
#define LN2F    0.6931471805599453f
#define LOG2EF  1.4426950408889634f

// ---------------------------------------------------------------------------
// fp32 -> f16 conversion, 8 elements per thread (sizes are multiples of 8):
// two b128 loads -> packed cvt -> one b128 store.
// ---------------------------------------------------------------------------
__global__ void cvt_f32_to_f16_x8(const float* __restrict__ src,
                                  _Float16* __restrict__ dst, int n) {
    int i = (blockIdx.x * blockDim.x + threadIdx.x) * 8;
    if (i < n) {
        float4 a = *(const float4*)(src + i);
        float4 b = *(const float4*)(src + i + 4);
        half8 h = {(_Float16)a.x, (_Float16)a.y, (_Float16)a.z, (_Float16)a.w,
                   (_Float16)b.x, (_Float16)b.y, (_Float16)b.z, (_Float16)b.w};
        *(half8*)(dst + i) = h;
    }
}

// ---------------------------------------------------------------------------
// shifted softplus via hardware transcendentals (v_exp_f32 / v_log_f32):
//   softplus(x) - ln2 = max(x,0) + ln2 * log2(1 + 2^(-|x|*log2e)) - ln2
// ---------------------------------------------------------------------------
__device__ __forceinline__ float softplus_m_ln2(float x) {
    float m  = fmaxf(x, 0.f);
    float t  = -fabsf(x) * LOG2EF;
    float u  = __builtin_amdgcn_exp2f(t);        // 2^t = e^{-|x|}
    float l2 = __builtin_amdgcn_logf(1.f + u);   // log2(1+u)
    return fmaf(l2, LN2F, m - LN2F);
}

// ---------------------------------------------------------------------------
// Main gather + GEMM + softplus + perm-reduce kernel.
// Block = 128 threads (4 waves). Each wave handles 4 sites (48 rows):
//   3 M-tiles x 4 N-tiles of v_wmma_f32_16x16x32_f16, K-loop of 16 steps.
// ---------------------------------------------------------------------------
__global__ __launch_bounds__(128, 1)
void lcnn_wmma_kernel(const _Float16* __restrict__ Xh,   // [N_SITES][64] f16
                      const _Float16* __restrict__ Wh,   // [64][512] f16 (row-major)
                      const int*      __restrict__ NS,   // [N_SITES][12][8]
                      const float*    __restrict__ bias, // [64]
                      float*          __restrict__ out)  // [N_SITES][64]
{
    __shared__ float red[4][48 * OUT_F];   // 12 KB per wave, 48 KB per block

    const int lane = threadIdx.x & 31;
    const int wave = threadIdx.x >> 5;
    const int n0   = (blockIdx.x * 4 + wave) * 4;   // first of this wave's 4 sites

    const int mrow   = lane & 15;          // row within M-tile / col within N-tile
    const int hi     = (lane >> 4) & 1;    // half-wave selector
    const int koff_h = hi ? 16 : 0;        // A-fragment K sub-offset (contiguous K order)

    // Precompute per-lane gather byte offsets: 3 M-tiles x 8 neighbors.
    unsigned aoff[3][8];
#pragma unroll
    for (int mt = 0; mt < 3; ++mt) {
        int r = mt * 16 + mrow;            // row 0..47 within wave
        int s = r / N_PERM, p = r % N_PERM;
        const int* nsp = NS + ((size_t)(n0 + s) * N_PERM + p) * N_NEIGH;
#pragma unroll
        for (int j = 0; j < N_NEIGH; ++j)
            aoff[mt][j] = (unsigned)nsp[j] * (unsigned)(NODE_F * sizeof(_Float16));
    }

    const v8f vzero = {0.f, 0.f, 0.f, 0.f, 0.f, 0.f, 0.f, 0.f};
    v8f acc[3][4];
#pragma unroll
    for (int mt = 0; mt < 3; ++mt)
#pragma unroll
        for (int nt = 0; nt < 4; ++nt)
            acc[mt][nt] = vzero;

    // ---- K loop: 16 steps of K=32 ----
#pragma unroll
    for (int ks = 0; ks < 16; ++ks) {
        const int k0  = ks * 32;
        const int j   = k0 >> 6;                 // neighbor index (2 ksteps per neighbor)
        const int kin = (k0 & 63) + koff_h;      // offset within neighbor vector (halves)

        // B fragments: lane = column n of B (= row of W, contiguous along K).
        // lanes 0-15 hold K=k0..k0+15, lanes 16-31 hold K=k0+16..k0+31.
        half16 bfrag[4];
#pragma unroll
        for (int nt = 0; nt < 4; ++nt) {
            const _Float16* bp = Wh + (size_t)(nt * 16 + mrow) * IN_F + k0 + (hi ? 16 : 0);
            bfrag[nt] = *(const half16*)bp;
        }

#pragma unroll
        for (int mt = 0; mt < 3; ++mt) {
            // A fragment: two contiguous 16B chunks at K=kin and K=kin+8.
            const _Float16* ap =
                (const _Float16*)((const char*)Xh + aoff[mt][j]) + kin;
            half8 alo = *(const half8*)ap;
            half8 ahi = *(const half8*)(ap + 8);
            half16 afrag = __builtin_shufflevector(
                alo, ahi, 0, 1, 2, 3, 4, 5, 6, 7, 8, 9, 10, 11, 12, 13, 14, 15);

#pragma unroll
            for (int nt = 0; nt < 4; ++nt)
                acc[mt][nt] = __builtin_amdgcn_wmma_f32_16x16x32_f16(
                    /*neg_a=*/false, afrag, /*neg_b=*/false, bfrag[nt],
                    /*c_mod=*/(short)0, acc[mt][nt],
                    /*reuse_a=*/false, /*reuse_b=*/false);
        }
    }

    // ---- Epilogue: bias + shifted softplus (hardware exp2/log2), stage to LDS ----
    float bv[4];
#pragma unroll
    for (int nt = 0; nt < 4; ++nt) bv[nt] = bias[nt * 16 + mrow];

#pragma unroll
    for (int mt = 0; mt < 3; ++mt)
#pragma unroll
        for (int nt = 0; nt < 4; ++nt)
#pragma unroll
            for (int g = 0; g < 8; ++g) {
                float x  = acc[mt][nt][g] + bv[nt];
                float sp = softplus_m_ln2(x);
                int row  = mt * 16 + hi * 8 + g;        // C layout: M = g + 8*hi
                int col  = nt * 16 + mrow;              // N = lane & 15
                red[wave][row * OUT_F + col] = sp;
            }

    __syncthreads();

    // ---- Reduce 12 perms per site; 256 outputs per wave, 8 per lane ----
#pragma unroll
    for (int it = 0; it < 8; ++it) {
        int o   = it * 32 + lane;     // 0..255
        int s   = o >> 6;             // site within wave (0..3)
        int col = o & 63;
        float sum = 0.f;
#pragma unroll
        for (int p = 0; p < N_PERM; ++p)
            sum += red[wave][(s * N_PERM + p) * OUT_F + col];
        out[(size_t)(n0 + s) * OUT_F + col] = sum;
    }
}

// ---------------------------------------------------------------------------
extern "C" void kernel_launch(void* const* d_in, const int* in_sizes, int n_in,
                              void* d_out, int out_size, void* d_ws, size_t ws_size,
                              hipStream_t stream) {
    const float* X  = (const float*)d_in[0];   // X_sites (50000,64) f32
    const int*   NS = (const int*)d_in[1];     // X_NSs   (50000,12,8) i32
    const float* W  = (const float*)d_in[2];   // W       (64,512) f32
    const float* b  = (const float*)d_in[3];   // b       (64,) f32
    float* out = (float*)d_out;                // (50000,64) f32

    // workspace layout: Xh (6,400,000 B) | Wh (65,536 B)
    _Float16* Xh = (_Float16*)d_ws;
    _Float16* Wh = (_Float16*)((char*)d_ws + (size_t)N_SITES * NODE_F * sizeof(_Float16));

    const int nx = N_SITES * NODE_F;           // 3,200,000 (multiple of 8)
    cvt_f32_to_f16_x8<<<(nx / 8 + 255) / 256, 256, 0, stream>>>(X, Xh, nx);
    const int nw = OUT_F * IN_F;               // 32,768 (multiple of 8)
    cvt_f32_to_f16_x8<<<(nw / 8 + 255) / 256, 256, 0, stream>>>(W, Wh, nw);

    // 3125 blocks x 4 waves x 4 sites = 50,000 sites exactly
    lcnn_wmma_kernel<<<N_SITES / 16, 128, 0, stream>>>(Xh, Wh, NS, b, out);
}
